// UGCA_DTI_32220844655403
// MI455X (gfx1250) — matrix operand order repaired and checked by reading, not compiled, not run on hardware
//
#include <hip/hip_runtime.h>
#include <math.h>

// ---------------------------------------------------------------------------
// UGCA-DTI forward for MI455X (gfx1250).
// Dense math on V_WMMA_F32_16X16X32_BF16 (bf16 in, f32 accumulate):
//   - aligned 64x64x32 block GEMM (no guards; float4 + ds_*_b128 only)
//   - general GEMM for odd shapes (clamped branch-free staging)
//   - flash cross-attention (hd=32 == WMMA K), float4-vectorized loads
// Cheap ops (LN, NIG gate, pooling) stay on fp32 VALU.
// ---------------------------------------------------------------------------

typedef __attribute__((ext_vector_type(16))) __bf16 v16bf;
typedef __attribute__((ext_vector_type(8)))  float  v8f;

#define D_MODEL 256
#define NHEAD   8
#define HDIM    32

#define BM 64
#define BN 64
#define BK 32
#define LDSS 40   // bf16 elements per LDS row (32 + 8 pad) -> 80B, 16B aligned

union Frag { v16bf v; uint4 q[2]; };

__device__ __forceinline__ float gelu_exact(float x) {
    return 0.5f * x * (1.0f + erff(x * 0.70710678118654752f));
}

__device__ __forceinline__ float softplusf(float x) {
    return (x > 20.0f) ? x : log1pf(__expf(x));
}

// pack two f32 -> dword of two bf16 (v_cvt_pk_bf16_f32)
__device__ __forceinline__ unsigned int pk2(float a, float b) {
    union { __bf16 h[2]; unsigned int u; } x;
    x.h[0] = (__bf16)a; x.h[1] = (__bf16)b;
    return x.u;
}

__device__ __forceinline__ uint4 pk8(float4 a, float4 b, float s) {
    uint4 r;
    r.x = pk2(a.x * s, a.y * s);
    r.y = pk2(a.z * s, a.w * s);
    r.z = pk2(b.x * s, b.y * s);
    r.w = pk2(b.z * s, b.w * s);
    return r;
}

// ---------------------------------------------------------------------------
// Shared compute core: given staged LDS tiles, do 4 WMMAs per wave.
// ---------------------------------------------------------------------------
__device__ __forceinline__ void gemm_tile_compute(const __bf16* As, const __bf16* Bs,
                                                  int wave, int llo, int lhi,
                                                  v8f acc[4]) {
    Frag fa;
    const __bf16* ap = &As[(wave * 16 + llo) * LDSS];
    fa.q[0] = *(const uint4*)&ap[lhi * 16];       // K = lhi*16 .. +7
    fa.q[1] = *(const uint4*)&ap[lhi * 16 + 8];   // K = lhi*16+8 .. +15
#pragma unroll
    for (int nt = 0; nt < 4; ++nt) {
        const __bf16* bp = &Bs[(nt * 16 + llo) * LDSS + lhi * 16];
        Frag fb;
        fb.q[0] = *(const uint4*)&bp[0];          // K = lhi*16 .. +7
        fb.q[1] = *(const uint4*)&bp[8];          // K = lhi*16+8 .. +15
        acc[nt] = __builtin_amdgcn_wmma_f32_16x16x32_bf16(
            false, fa.v, false, fb.v, (short)0, acc[nt], false, false);
    }
}

// ---------------------------------------------------------------------------
// ALIGNED GEMM: requires M%64==0, N%64==0, K%32==0. No guards at all.
// Block = 128 threads (4 waves), tile = 64x64, K-step 32.
// ---------------------------------------------------------------------------
__global__ void __launch_bounds__(128)
gemm_aligned_kernel(const float* __restrict__ A,
                    const float* __restrict__ W,
                    const float* __restrict__ bias,
                    float* __restrict__ C,
                    int M, int N, int K, int act) {
    __shared__ __bf16 As[BM * LDSS];
    __shared__ __bf16 Bs[BN * LDSS];

    const int tid  = threadIdx.x;
    const int lane = tid & 31;
    const int wave = tid >> 5;
    const int llo  = lane & 15;
    const int lhi  = lane >> 4;
    const int m0   = blockIdx.y * BM;
    const int n0   = blockIdx.x * BN;

    const int arow = tid >> 1;           // 0..63
    const int akh  = (tid & 1) * 16;     // 0|16
    const int bk   = tid >> 2;           // 0..31
    const int bnq  = (tid & 3) * 16;     // 0,16,32,48

    const float* Ap = A + (size_t)(m0 + arow) * K + akh;
    const float* Wp = W + (size_t)bk * N + n0 + bnq;
    uint4* adst = (uint4*)&As[arow * LDSS + akh];

    v8f acc[4] = {{}, {}, {}, {}};

    for (int k0 = 0; k0 < K; k0 += BK) {
        {   // stage A (BM x BK), pure float4 + b128
            const float4* src = (const float4*)(Ap + k0);
            float4 f0 = src[0], f1 = src[1], f2 = src[2], f3 = src[3];
            adst[0] = pk8(f0, f1, 1.0f);
            adst[1] = pk8(f2, f3, 1.0f);
            if (k0 + BK < K)
                __builtin_prefetch(Ap + k0 + BK, 0, 1);
        }
        {   // stage B transposed: Bs[col][k]
            const float4* src = (const float4*)(Wp + (size_t)k0 * N);
            float4 f0 = src[0], f1 = src[1], f2 = src[2], f3 = src[3];
            float v[16] = { f0.x, f0.y, f0.z, f0.w, f1.x, f1.y, f1.z, f1.w,
                            f2.x, f2.y, f2.z, f2.w, f3.x, f3.y, f3.z, f3.w };
#pragma unroll
            for (int j = 0; j < 16; ++j)
                Bs[(bnq + j) * LDSS + bk] = (__bf16)v[j];
            if (k0 + BK < K)
                __builtin_prefetch(Wp + (size_t)(k0 + BK) * N, 0, 1);
        }
        __syncthreads();
        gemm_tile_compute(As, Bs, wave, llo, lhi, acc);
        __syncthreads();
    }

#pragma unroll
    for (int nt = 0; nt < 4; ++nt) {
        const int col = n0 + nt * 16 + llo;
        const float bv = bias ? bias[col] : 0.0f;
#pragma unroll
        for (int r = 0; r < 8; ++r) {
            const int m = m0 + wave * 16 + r + 8 * lhi;
            float v = acc[nt][r] + bv;
            if (act == 1) v = gelu_exact(v);
            C[(size_t)m * N + col] = v;
        }
    }
}

// ---------------------------------------------------------------------------
// GENERAL GEMM: any M%16==0, any N, any K. Staging uses clamped addresses +
// zero-selects (branch-free; no exec-mask per-element loads).
// ---------------------------------------------------------------------------
__global__ void __launch_bounds__(128)
gemm_general_kernel(const float* __restrict__ A,
                    const float* __restrict__ W,
                    const float* __restrict__ bias,
                    float* __restrict__ C,
                    int M, int N, int K, int act) {
    __shared__ __bf16 As[BM * LDSS];
    __shared__ __bf16 Bs[BN * LDSS];

    const int tid  = threadIdx.x;
    const int lane = tid & 31;
    const int wave = tid >> 5;
    const int llo  = lane & 15;
    const int lhi  = lane >> 4;
    const int m0   = blockIdx.y * BM;
    const int n0   = blockIdx.x * BN;

    const int arow = tid >> 1;
    const int akh  = (tid & 1) * 16;
    const int bk   = tid >> 2;
    const int bnq  = (tid & 3) * 16;

    const int gm = m0 + arow;
    const float* Arow_p = A + (size_t)min(gm, M - 1) * K;
    const bool mok = (gm < M);

    v8f acc[4] = {{}, {}, {}, {}};

    for (int k0 = 0; k0 < K; k0 += BK) {
        {   // stage A: clamped unconditional loads, zero-select OOB
            float v[16];
#pragma unroll
            for (int j = 0; j < 16; ++j) {
                int kk = k0 + akh + j;
                float x = Arow_p[min(kk, K - 1)];
                v[j] = (mok && kk < K) ? x : 0.0f;
            }
            uint4* dst = (uint4*)&As[arow * LDSS + akh];
            uint4 lo, hi;
            lo.x = pk2(v[0], v[1]);   lo.y = pk2(v[2], v[3]);
            lo.z = pk2(v[4], v[5]);   lo.w = pk2(v[6], v[7]);
            hi.x = pk2(v[8], v[9]);   hi.y = pk2(v[10], v[11]);
            hi.z = pk2(v[12], v[13]); hi.w = pk2(v[14], v[15]);
            dst[0] = lo;
            dst[1] = hi;
        }
        {   // stage B transposed with clamped loads
            const int gk = k0 + bk;
            const float* Wrow = W + (size_t)min(gk, K - 1) * N;
            const bool kok = (gk < K);
#pragma unroll
            for (int j = 0; j < 16; ++j) {
                int col = n0 + bnq + j;
                float x = Wrow[min(col, N - 1)];
                Bs[(bnq + j) * LDSS + bk] = (__bf16)((kok && col < N) ? x : 0.0f);
            }
        }
        __syncthreads();
        gemm_tile_compute(As, Bs, wave, llo, lhi, acc);
        __syncthreads();
    }

#pragma unroll
    for (int nt = 0; nt < 4; ++nt) {
        const int col = n0 + nt * 16 + llo;
        if (col >= N) continue;
        const float bv = bias ? bias[col] : 0.0f;
#pragma unroll
        for (int r = 0; r < 8; ++r) {
            const int m = m0 + wave * 16 + r + 8 * lhi;
            if (m < M) {
                float v = acc[nt][r] + bv;
                if (act == 1) v = gelu_exact(v);
                C[(size_t)m * N + col] = v;
            }
        }
    }
}

// ---------------------------------------------------------------------------
// Flash cross-attention, one wave per (batch, head, 16 query rows).
// hd = 32 = WMMA K dim. Scores get log(gate+1e-9) per key and -1e9 where
// mask==0; online softmax; P tile re-laid out C->A through LDS.
// grid = (Lq/16, NHEAD, B), block = 32.
// ---------------------------------------------------------------------------
__global__ void __launch_bounds__(32)
flash_xattn_kernel(const float* __restrict__ Q,
                   const float* __restrict__ Km,
                   const float* __restrict__ Vm,
                   const float* __restrict__ gate,
                   const int*   __restrict__ mask,
                   float* __restrict__ Out,
                   int Lq, int Lk) {
    __shared__ float pst[16 * 32];
    const int lane = threadIdx.x & 31;
    const int llo  = lane & 15;
    const int lhi  = lane >> 4;
    const int qt = blockIdx.x, hh = blockIdx.y, b = blockIdx.z;
    const float scale = 0.17677669529663687f; // 1/sqrt(32)

    const float* Qb = Q  + ((size_t)b * Lq) * D_MODEL + hh * HDIM;
    const float* Kb = Km + ((size_t)b * Lk) * D_MODEL + hh * HDIM;
    const float* Vb = Vm + ((size_t)b * Lk) * D_MODEL + hh * HDIM;
    const float* gb = gate + (size_t)b * Lk;
    const int*   mb = mask + (size_t)b * Lk;

    Frag fq;
    {
        const int qrow = qt * 16 + llo;
        const float4* p0 = (const float4*)(Qb + (size_t)qrow * D_MODEL + lhi * 16);
        const float4* p1 = (const float4*)(Qb + (size_t)qrow * D_MODEL + lhi * 16 + 8);
        fq.q[0] = pk8(p0[0], p0[1], scale);
        fq.q[1] = pk8(p1[0], p1[1], scale);
    }

    float m_run[8], l_run[8];
    v8f o0 = {}, o1 = {};
#pragma unroll
    for (int r = 0; r < 8; ++r) { m_run[r] = -3.0e38f; l_run[r] = 0.0f; }

    for (int kb = 0; kb < Lk; kb += 32) {
        const int key0 = kb + llo, key1 = kb + 16 + llo;
        Frag fk0, fk1;
        {
            const float4* p = (const float4*)(Kb + (size_t)key0 * D_MODEL + lhi * 16);
            fk0.q[0] = pk8(p[0], p[1], 1.0f);
            fk0.q[1] = pk8(p[2], p[3], 1.0f);
            const float4* q = (const float4*)(Kb + (size_t)key1 * D_MODEL + lhi * 16);
            fk1.q[0] = pk8(q[0], q[1], 1.0f);
            fk1.q[1] = pk8(q[2], q[3], 1.0f);
        }
        v8f zero = {};
        v8f s0 = __builtin_amdgcn_wmma_f32_16x16x32_bf16(
            false, fq.v, false, fk0.v, (short)0, zero, false, false);
        v8f s1 = __builtin_amdgcn_wmma_f32_16x16x32_bf16(
            false, fq.v, false, fk1.v, (short)0, zero, false, false);

        const float lg0 = __logf(gb[key0] + 1e-9f);
        const float lg1 = __logf(gb[key1] + 1e-9f);
        const bool ok0 = (mb[key0] != 0), ok1 = (mb[key1] != 0);

#pragma unroll
        for (int r = 0; r < 8; ++r) {
            float v0 = ok0 ? (s0[r] + lg0) : -1e9f;
            float v1 = ok1 ? (s1[r] + lg1) : -1e9f;
            float rm = fmaxf(v0, v1);
            for (int off = 1; off < 16; off <<= 1)
                rm = fmaxf(rm, __shfl_xor(rm, off, 32));
            float mnew = fmaxf(m_run[r], rm);
            float corr = __expf(m_run[r] - mnew);
            float p0 = __expf(v0 - mnew);
            float p1 = __expf(v1 - mnew);
            float ps = p0 + p1;
            for (int off = 1; off < 16; off <<= 1)
                ps += __shfl_xor(ps, off, 32);
            l_run[r] = l_run[r] * corr + ps;
            m_run[r] = mnew;
            o0[r] *= corr;
            o1[r] *= corr;
            pst[(r + 8 * lhi) * 32 + llo]      = p0;
            pst[(r + 8 * lhi) * 32 + 16 + llo] = p1;
        }
        __syncthreads();

        Frag fp;
        {
            const float4* p0 = (const float4*)&pst[llo * 32 + lhi * 16];
            const float4* p1 = (const float4*)&pst[llo * 32 + lhi * 16 + 8];
            fp.q[0] = pk8(p0[0], p0[1], 1.0f);
            fp.q[1] = pk8(p1[0], p1[1], 1.0f);
        }
        Frag fv0, fv1;
#pragma unroll
        for (int h = 0; h < 16; ++h) {
            int kv = kb + lhi * 16 + h;
            fv0.v[h] = (__bf16)Vb[(size_t)kv * D_MODEL + llo];
            fv1.v[h] = (__bf16)Vb[(size_t)kv * D_MODEL + 16 + llo];
        }
        o0 = __builtin_amdgcn_wmma_f32_16x16x32_bf16(
            false, fp.v, false, fv0.v, (short)0, o0, false, false);
        o1 = __builtin_amdgcn_wmma_f32_16x16x32_bf16(
            false, fp.v, false, fv1.v, (short)0, o1, false, false);
        __syncthreads();
    }

    float* Ob = Out + ((size_t)b * Lq) * D_MODEL + hh * HDIM;
#pragma unroll
    for (int r = 0; r < 8; ++r) {
        int qr = qt * 16 + r + 8 * lhi;
        float inv = 1.0f / l_run[r];
        Ob[(size_t)qr * D_MODEL + llo]      = o0[r] * inv;
        Ob[(size_t)qr * D_MODEL + 16 + llo] = o1[r] * inv;
    }
}

// ---------------------------------------------------------------------------
// NIG uncertainty gate
// ---------------------------------------------------------------------------
__global__ void __launch_bounds__(32)
ugate_kernel(const float* __restrict__ X,
             const float* __restrict__ Wg,
             const float* __restrict__ bg,
             float* __restrict__ gate, int T) {
    int t = blockIdx.x;
    if (t >= T) return;
    int lane = threadIdx.x & 31;
    const float* x = X + (size_t)t * D_MODEL;
    float n[4] = {0.f, 0.f, 0.f, 0.f};
    for (int d = lane; d < D_MODEL; d += 32) {
        float xv = x[d];
        const float4 w = *(const float4*)(Wg + d * 4);
        n[0] += xv * w.x; n[1] += xv * w.y; n[2] += xv * w.z; n[3] += xv * w.w;
    }
#pragma unroll
    for (int j = 0; j < 4; ++j)
        for (int off = 1; off < 32; off <<= 1)
            n[j] += __shfl_xor(n[j], off, 32);
    if (lane == 0) {
        float mu    = n[0] + bg[0];
        float v     = softplusf(n[1] + bg[1]) + 1e-6f;
        float alpha = softplusf(n[2] + bg[2]) + 1.0f + 1e-6f;
        float beta  = softplusf(n[3] + bg[3]) + 1e-6f;
        float sig   = beta / (v * (alpha - 1.0f));
        float g = (1.0f / (1.0f + __expf(-mu))) * __expf(-sig);
        gate[t] = fminf(fmaxf(g, 1e-3f), 1.0f);
    }
}

// ---------------------------------------------------------------------------
// Out = act(LN(X + Res) * g + b); Res may be null; act=1 -> GELU after affine
// ---------------------------------------------------------------------------
__global__ void __launch_bounds__(32)
ln_kernel(const float* __restrict__ X,
          const float* __restrict__ Res,
          const float* __restrict__ gam,
          const float* __restrict__ bet,
          float* __restrict__ Out,
          int Dm, int act) {
    int t = blockIdx.x;
    int lane = threadIdx.x & 31;
    float vals[32];
    const int nper = Dm / 32;
    const float* x = X + (size_t)t * Dm;
    const float* r = Res ? (Res + (size_t)t * Dm) : nullptr;
    float s = 0.0f;
    for (int i = 0; i < nper; ++i) {
        int d = lane + i * 32;
        float v = x[d] + (r ? r[d] : 0.0f);
        vals[i] = v;
        s += v;
    }
    for (int off = 1; off < 32; off <<= 1) s += __shfl_xor(s, off, 32);
    float mean = s / (float)Dm;
    float sq = 0.0f;
    for (int i = 0; i < nper; ++i) { float d = vals[i] - mean; sq += d * d; }
    for (int off = 1; off < 32; off <<= 1) sq += __shfl_xor(sq, off, 32);
    float inv = rsqrtf(sq / (float)Dm + 1e-5f);
    float* o = Out + (size_t)t * Dm;
    for (int i = 0; i < nper; ++i) {
        int d = lane + i * 32;
        float v = (vals[i] - mean) * inv * gam[d] + bet[d];
        if (act == 1) v = gelu_exact(v);
        o[d] = v;
    }
}

// ---------------------------------------------------------------------------
// Masked mean/max pool: X [B,L,256] -> Z2 [B,512] = [mean | max]
// ---------------------------------------------------------------------------
__global__ void pool_kernel(const float* __restrict__ X,
                            const int* __restrict__ mask,
                            float* __restrict__ Z2, int L) {
    int idx = blockIdx.x * blockDim.x + threadIdx.x;  // b*256 + d
    int b = idx >> 8, d = idx & 255;
    const float* xb = X + ((size_t)b * L) * D_MODEL + d;
    const int* mb = mask + (size_t)b * L;
    float s = 0.0f, mx = -1e9f, cnt = 0.0f;
    for (int l = 0; l < L; ++l) {
        float v = xb[(size_t)l * D_MODEL];
        int m = mb[l];
        s += v * (float)m;
        cnt += (float)m;
        mx = fmaxf(mx, (m == 0) ? -1e9f : v);
    }
    Z2[(size_t)b * 512 + d]       = s / fmaxf(cnt, 1e-9f);
    Z2[(size_t)b * 512 + 256 + d] = mx;
}

__global__ void concat2_kernel(const float* __restrict__ A,
                               const float* __restrict__ B2,
                               float* __restrict__ Out) {
    int i = blockIdx.x * blockDim.x + threadIdx.x;    // B*512
    int b = i >> 9, j = i & 511;
    Out[i] = (j < 256) ? A[b * 256 + j] : B2[b * 256 + (j - 256)];
}

__global__ void matchfeat_kernel(const float* __restrict__ fd,
                                 const float* __restrict__ fp,
                                 float* __restrict__ Out) {
    int i = blockIdx.x * blockDim.x + threadIdx.x;    // B*1024
    int b = i >> 10, j = i & 1023;
    int d = j & 255, sec = j >> 8;
    float a = fd[b * 256 + d], c = fp[b * 256 + d];
    float v = (sec == 0) ? a : (sec == 1) ? c : (sec == 2) ? a * c : fabsf(a - c);
    Out[i] = v;
}

// ---------------------------------------------------------------------------
// Host-side orchestration. Input order: molclr, esm2, chemberta, gvp,
// mask_d, mask_p, then params flattened JAX-style (dict keys sorted, lists in
// order), Linear={W,b}, LayerNorm={b,g}.
// ---------------------------------------------------------------------------
extern "C" void kernel_launch(void* const* d_in, const int* in_sizes, int n_in,
                              void* d_out, int out_size, void* d_ws, size_t ws_size,
                              hipStream_t stream) {
    (void)in_sizes; (void)n_in; (void)out_size; (void)ws_size;
    const int B = 32, Ld = 128, Lp = 1024;
    const int Td = B * Ld;        // 4096 drug tokens
    const int Tp = B * Lp;        // 32768 protein tokens

    const float* molclr    = (const float*)d_in[0];
    const float* esm2      = (const float*)d_in[1];
    const float* chemberta = (const float*)d_in[2];
    const float* gvp       = (const float*)d_in[3];
    const int*   mask_d    = (const int*)d_in[4];
    const int*   mask_p    = (const int*)d_in[5];

    auto P = [&](int i) { return (const float*)d_in[i]; };
    const int LBASE = 24;
    auto LP = [&](int l, int off) { return (const float*)d_in[LBASE + l * 36 + off]; };
    const int PB = LBASE + 3 * 36;  // 132: pool.W

    float* ws = (float*)d_ws;
    size_t off = 0;
    auto take = [&](size_t n) { float* p = ws + off; off += n; return p; };
    float* Hd  = take((size_t)Td * 256);
    float* Hp  = take((size_t)Tp * 256);
    float* Xq  = take((size_t)Tp * 256);
    float* Xk  = take((size_t)Tp * 256);
    float* Xv  = take((size_t)Tp * 256);
    float* Xa  = take((size_t)Tp * 256);
    float* Ft  = take((size_t)Tp * 1024);
    float* gd  = take((size_t)Td);
    float* gp  = take((size_t)Tp);
    float* z2  = take((size_t)B * 512);
    float* zd  = take((size_t)B * 256);
    float* zp  = take((size_t)B * 256);
    float* cc  = take((size_t)B * 256);
    float* gg  = take((size_t)B * 256);
    float* cat = take((size_t)B * 512);
    float* fdv = take((size_t)B * 256);
    float* fpv = take((size_t)B * 256);
    float* t1  = take((size_t)B * 1024);
    float* t2  = take((size_t)B * 1024);

    auto GEMM = [&](const float* A, const float* W, const float* bias, float* C,
                    int M, int N, int K, int act) {
        dim3 grid((N + BN - 1) / BN, (M + BM - 1) / BM);
        if ((M % BM) == 0 && (N % BN) == 0 && (K % BK) == 0)
            gemm_aligned_kernel<<<grid, 128, 0, stream>>>(A, W, bias, C, M, N, K, act);
        else
            gemm_general_kernel<<<grid, 128, 0, stream>>>(A, W, bias, C, M, N, K, act);
    };

    // ---- input projections ----
    GEMM(molclr,    P(PB + 4), P(PB + 5), Hd, Td, 256, 300, 0);  // proj_d (general: K=300)
    GEMM(esm2,      P(PB + 8), P(PB + 9), Hp, Tp, 256, 1280, 0); // proj_p (aligned)
    GEMM(chemberta, P(PB + 2), P(PB + 3), cc, B, 256, 384, 0);   // proj_c (general: M=32)
    GEMM(gvp,       P(PB + 6), P(PB + 7), gg, B, 256, 256, 0);   // proj_g (general: M=32)

    // ---- UGCA layers ----
    for (int l = 0; l < 3; ++l) {
        ugate_kernel<<<Td, 32, 0, stream>>>(Hd, LP(l, 8),  LP(l, 9),  gd, Td);
        ugate_kernel<<<Tp, 32, 0, stream>>>(Hp, LP(l, 10), LP(l, 11), gp, Tp);

        // a_d = xattn(Hd@wqd, Hp@wkp, Hp@wvp, g_p, mask_p)
        GEMM(Hd, LP(l, 28), LP(l, 29), Xq, Td, 256, 256, 0);
        GEMM(Hp, LP(l, 26), LP(l, 27), Xk, Tp, 256, 256, 0);
        GEMM(Hp, LP(l, 34), LP(l, 35), Xv, Tp, 256, 256, 0);
        {
            dim3 g(Ld / 16, NHEAD, B);
            flash_xattn_kernel<<<g, 32, 0, stream>>>(Xq, Xk, Xv, gp, mask_p, Xa, Ld, Lp);
        }
        GEMM(Xa, LP(l, 20), LP(l, 21), Xq, Td, 256, 256, 0);          // o_d
        ln_kernel<<<Td, 32, 0, stream>>>(Xq, Hd, LP(l, 13), LP(l, 12), Hd, 256, 0);

        // a_p = xattn(Hp@wqp, Hd@wkd, Hd@wvd, g_d, mask_d)  (updated Hd)
        GEMM(Hp, LP(l, 30), LP(l, 31), Xq, Tp, 256, 256, 0);
        GEMM(Hd, LP(l, 24), LP(l, 25), Xk, Td, 256, 256, 0);
        GEMM(Hd, LP(l, 32), LP(l, 33), Xv, Td, 256, 256, 0);
        {
            dim3 g(Lp / 16, NHEAD, B);
            flash_xattn_kernel<<<g, 32, 0, stream>>>(Xq, Xk, Xv, gd, mask_d, Xa, Lp, Ld);
        }
        GEMM(Xa, LP(l, 22), LP(l, 23), Xq, Tp, 256, 256, 0);          // o_p
        ln_kernel<<<Tp, 32, 0, stream>>>(Xq, Hp, LP(l, 17), LP(l, 16), Hp, 256, 0);

        // FFN drug
        GEMM(Hd, LP(l, 0), LP(l, 1), Ft, Td, 1024, 256, 1);
        GEMM(Ft, LP(l, 2), LP(l, 3), Xq, Td, 256, 1024, 0);
        ln_kernel<<<Td, 32, 0, stream>>>(Xq, Hd, LP(l, 15), LP(l, 14), Hd, 256, 0);
        // FFN protein
        GEMM(Hp, LP(l, 4), LP(l, 5), Ft, Tp, 1024, 256, 1);
        GEMM(Ft, LP(l, 6), LP(l, 7), Xq, Tp, 256, 1024, 0);
        ln_kernel<<<Tp, 32, 0, stream>>>(Xq, Hp, LP(l, 19), LP(l, 18), Hp, 256, 0);
    }

    // ---- pooling ----
    pool_kernel<<<(B * 256) / 256, 256, 0, stream>>>(Hd, mask_d, z2, Ld);
    GEMM(z2, P(PB + 0), P(PB + 1), zd, B, 256, 512, 0);
    pool_kernel<<<(B * 256) / 256, 256, 0, stream>>>(Hp, mask_p, z2, Lp);
    GEMM(z2, P(PB + 0), P(PB + 1), zp, B, 256, 512, 0);

    // ---- fusion ----
    concat2_kernel<<<(B * 512) / 256, 256, 0, stream>>>(zd, cc, cat);
    GEMM(cat, P(6), P(7), t1, B, 256, 512, 1);    // fus_d1 + gelu
    GEMM(t1, P(8), P(9), fdv, B, 256, 256, 0);    // fus_d2
    concat2_kernel<<<(B * 512) / 256, 256, 0, stream>>>(zp, gg, cat);
    GEMM(cat, P(10), P(11), t1, B, 256, 512, 1);  // fus_p1 + gelu
    GEMM(t1, P(12), P(13), fpv, B, 256, 256, 0);  // fus_p2

    // ---- MatchMLP head ----
    matchfeat_kernel<<<(B * 1024) / 256, 256, 0, stream>>>(fdv, fpv, t1);
    GEMM(t1, P(14), P(15), t2, B, 1024, 1024, 0);                 // head_l1
    ln_kernel<<<B, 32, 0, stream>>>(t2, nullptr, P(21), P(20), t1, 1024, 1);
    GEMM(t1, P(16), P(17), t2, B, 512, 1024, 0);                  // head_l2
    ln_kernel<<<B, 32, 0, stream>>>(t2, nullptr, P(23), P(22), t1, 512, 1);
    GEMM(t1, P(18), P(19), (float*)d_out, B, 1, 512, 0);          // head_l3
}
